// MambaBlock_489626272162
// MI455X (gfx1250) — matrix operand order repaired and checked by reading, not compiled, not run on hardware
//
#include <hip/hip_runtime.h>
#include <hip/hip_bf16.h>

typedef __attribute__((ext_vector_type(16))) __bf16 v16bf;
typedef __attribute__((ext_vector_type(8)))  float  v8f;

__device__ __forceinline__ float silu_f(float x) {
    return x / (1.0f + __expf(-x));
}
__device__ __forceinline__ float softplus_f(float x) {
    return (x > 20.0f) ? x : __logf(1.0f + __expf(x));
}

// A-fragment (16xK tile, 16-bit elems): lanes 0-15 hold M=0..15 with K in {0..15},
// lanes 16-31 hold M=0..15 with K in {16..31}. Source: f32 row-major [M][ld].
__device__ __forceinline__ v16bf load_a_f32(const float* m, int ld, int kbase) {
    const int lane = threadIdx.x & 31;
    const int M    = lane & 15;
    const int khi  = (lane < 16) ? 0 : 16;
    v16bf a;
#pragma unroll
    for (int i = 0; i < 16; ++i) {
        int K = kbase + khi + i;   // K=i (plus khi)
        a[i] = (__bf16)m[M * ld + K];
    }
    return a;
}

// B-fragment (Kx16 tile): lane = N (mod 16), lanes 0-15 hold K=kbase..+15,
// lanes 16-31 hold K=kbase+16..+31. Source: f32 global, row-major [K][ld]. UNCONDITIONAL.
__device__ __forceinline__ v16bf load_b_g_full(const float* m, int ld, int kbase, int nbase) {
    const int lane = threadIdx.x & 31;
    const int n    = (lane & 15) + nbase;
    const int kblk = (lane < 16) ? 0 : 16;
    v16bf b;
#pragma unroll
    for (int i = 0; i < 16; ++i) {
        int K = kbase + kblk + i;
        b[i] = (__bf16)m[K * ld + n];
    }
    return b;
}

// B-fragment from a pre-converted bf16 LDS buffer (already zero-padded). No cvt needed.
__device__ __forceinline__ v16bf load_b_ldsbf(const __bf16* m, int ld, int kbase, int nbase) {
    const int lane = threadIdx.x & 31;
    const int n    = (lane & 15) + nbase;
    const int kblk = (lane < 16) ? 0 : 16;
    v16bf b;
#pragma unroll
    for (int i = 0; i < 16; ++i) {
        int K = kbase + kblk + i;
        b[i] = m[K * ld + n];
    }
    return b;
}

// D-fragment (16x16 f32): VGPR r holds M=r (lanes 0-15) / M=r+8 (lanes 16-31), N=lane%16.
__device__ __forceinline__ void store_d_lds(float* m, int ld, int nbase, v8f d) {
    const int lane = threadIdx.x & 31;
    const int N    = (lane & 15) + nbase;
    const int Mb   = (lane < 16) ? 0 : 8;
#pragma unroll
    for (int r = 0; r < 8; ++r) m[(Mb + r) * ld + N] = d[r];
}

#define WMMA_BF16(a, b, c) \
    __builtin_amdgcn_wmma_f32_16x16x32_bf16(false, (a), false, (b), (short)0, (c), false, false)

// One workgroup (8 wave32s) per (batch, group): 512 sequential timesteps, tiled by 16.
__global__ __launch_bounds__(256) void mamba_fused_scan(
    const float* __restrict__ x,      // [8][4096][64]
    const float* __restrict__ latent, // [8][8][128][16]
    const float* __restrict__ inw,    // [64][256]
    const float* __restrict__ convw,  // [128][1][4]
    const float* __restrict__ convb,  // [128]
    const float* __restrict__ xpw,    // [128][36]
    const float* __restrict__ dtw,    // [4][128]
    const float* __restrict__ dtb,    // [128]
    const float* __restrict__ outw,   // [128][64]
    const float* __restrict__ Alog,   // [128][16]
    const float* __restrict__ Dp,     // [128]
    float* __restrict__ out,          // [8][4096][64]
    float* __restrict__ hout)         // [64][128][16] scratch for fft kernel
{
    __shared__ float  s_dtw[4 * 128];
    __shared__ float  s_dtb[128];
    __shared__ float  s_convw[128 * 4];
    __shared__ float  s_convb[128];
    __shared__ float  s_D[128];
    __shared__ __bf16 s_xpw[128 * 48];  // x_proj_w, bf16, zero-padded 36 -> 48 cols
    __shared__ float  s_xr[16 * 256];   // xi | res
    __shared__ float  s_u[16 * 128];    // u, later reused as y
    __shared__ float  s_xdbl[16 * 48];  // delta_raw(4) | B(16) | C(16) | pad
    __shared__ float  s_delta[16 * 128];
    __shared__ float  s_y[16 * 128];    // y accumulator (ds_add_f32, 2 contenders)
    __shared__ float  s_tail[3 * 128];  // conv history (xi rows t-3..t-1)

    const int tid  = threadIdx.x;
    const int bg   = blockIdx.x;
    const int b    = bg >> 3;
    const int g    = bg & 7;
    const int wave = tid >> 5;

    // ---- preload small constants ----
    for (int i = tid; i < 512; i += 256) { s_dtw[i] = dtw[i]; s_convw[i] = convw[i]; }
    if (tid < 128) { s_dtb[tid] = dtb[tid]; s_convb[tid] = convb[tid]; s_D[tid] = Dp[tid]; }
    // pre-convert x_proj_w to padded bf16
    for (int i = tid; i < 128 * 48; i += 256) {
        int r = i / 48, cc = i - r * 48;
        float v = (cc < 36) ? xpw[r * 36 + cc] : 0.0f;
        s_xpw[i] = (__bf16)v;
    }
    // zero the y accumulator
    for (int i = tid; i < 2048; i += 256) s_y[i] = 0.0f;

    // ---- per-thread recurrence state: thread owns (d, n in [nb,nb+8)) ----
    const int d_s = tid & 127;
    const int nb  = (tid >> 7) << 3;
    float h[8], Arow[8];
#pragma unroll
    for (int j = 0; j < 8; ++j) {
        Arow[j] = -__expf(Alog[d_s * 16 + nb + j]);
        h[j]    = latent[(size_t)bg * 2048 + d_s * 16 + nb + j];
    }

    const size_t brow = (size_t)b * 4096;
    const int    t0g  = g * 512;

    // ---- initial conv tail: xi for rows t0g-3..t0g-1 via direct matvec ----
    for (int i = tid; i < 384; i += 256) {
        int r = i >> 7, d = i & 127;
        int t = t0g - 3 + r;
        float acc = 0.0f;
        if (t >= 0) {
            const float* xrow = x + (brow + t) * 64;
#pragma unroll
            for (int k = 0; k < 64; ++k) acc += xrow[k] * inw[k * 256 + d];
        }
        s_tail[r * 128 + d] = acc;
    }
    __syncthreads();

    for (int c = 0; c < 32; ++c) {
        const int trow = t0g + c * 16;

        // stage 1: xr = X @ in_proj (16x64 @ 64x256); A straight from global x rows,
        // two K-slices hoisted and reused across this wave's two output tiles.
        {
            const float* xrow = x + (brow + trow) * 64;
            v16bf a0 = load_a_f32(xrow, 64, 0);
            v16bf a1 = load_a_f32(xrow, 64, 32);
#pragma unroll
            for (int rep = 0; rep < 2; ++rep) {
                const int nbase = (wave + rep * 8) * 16;
                v8f acc = {};
                acc = WMMA_BF16(a0, load_b_g_full(inw, 256, 0, nbase), acc);
                acc = WMMA_BF16(a1, load_b_g_full(inw, 256, 32, nbase), acc);
                store_d_lds(s_xr, 256, nbase, acc);
            }
        }
        __syncthreads();

        // stage 2: causal depthwise conv (width 4) + SiLU -> u
        for (int i = tid * 8, e = tid * 8 + 8; i < e; ++i) {
            int t = i >> 7, d = i & 127;
            float acc = s_convb[d];
#pragma unroll
            for (int k = 0; k < 4; ++k) {
                int r = t - 3 + k;
                float xv = (r >= 0) ? s_xr[r * 256 + d] : s_tail[(3 + r) * 128 + d];
                acc += s_convw[d * 4 + k] * xv;
            }
            s_u[t * 128 + d] = silu_f(acc);
        }
        __syncthreads();

        // stage 3: x_dbl = U @ x_proj (16x128 @ 128x48 padded, bf16 weights in LDS)
        if (wave < 3) {
            const int nbase = wave * 16;
            v8f acc = {};
#pragma unroll
            for (int kb = 0; kb < 128; kb += 32) {
                v16bf a  = load_a_f32(s_u, 128, kb);
                v16bf bf = load_b_ldsbf(s_xpw, 48, kb, nbase);
                acc = WMMA_BF16(a, bf, acc);
            }
            store_d_lds(s_xdbl, 48, nbase, acc);
        }
        __syncthreads();

        // stage 4: delta = softplus(raw @ dt_proj + bias)
        for (int i = tid * 8, e = tid * 8 + 8; i < e; ++i) {
            int t = i >> 7, d = i & 127;
            float acc = s_dtb[d];
#pragma unroll
            for (int r = 0; r < 4; ++r) acc += s_xdbl[t * 48 + r] * s_dtw[r * 128 + d];
            s_delta[t * 128 + d] = softplus_f(acc);
        }
        __syncthreads();

        // stage 5: 16 sequential recurrence steps, state in registers
#pragma unroll 1
        for (int t = 0; t < 16; ++t) {
            float dlt = s_delta[t * 128 + d_s];
            float du  = dlt * s_u[t * 128 + d_s];
            float acc = 0.0f;
#pragma unroll
            for (int j = 0; j < 8; ++j) {
                float Bv = s_xdbl[t * 48 + 4  + nb + j];
                float Cv = s_xdbl[t * 48 + 20 + nb + j];
                float dA = __expf(dlt * Arow[j]);
                h[j] = dA * h[j] + du * Bv;
                acc += h[j] * Cv;
            }
            atomicAdd(&s_y[t * 128 + d_s], acc);   // ds_add_f32, 2 contenders per slot
        }
        __syncthreads();

        // stage 6: y = (sum_n hC + u*D) * silu(res); reuse s_u as y; re-zero s_y slot
        for (int i = tid * 8, e = tid * 8 + 8; i < e; ++i) {
            int t = i >> 7, d = i & 127;
            float ysum = s_y[t * 128 + d];
            s_y[t * 128 + d] = 0.0f;               // re-arm accumulator for next chunk
            float y = ysum + s_u[t * 128 + d] * s_D[d];
            y *= silu_f(s_xr[t * 256 + 128 + d]);
            s_u[t * 128 + d] = y;
        }
        // update conv tail from last 3 xi rows of this chunk
        for (int i = tid; i < 384; i += 256) {
            int r = i >> 7, d = i & 127;
            s_tail[r * 128 + d] = s_xr[(13 + r) * 256 + d];
        }
        __syncthreads();

        // stage 7: out = Y @ out_proj (16x128 @ 128x64), D-fragments straight to global
        if (wave < 4) {
            const int nbase = wave * 16;
            v8f acc = {};
#pragma unroll
            for (int kb = 0; kb < 128; kb += 32) {
                v16bf a  = load_a_f32(s_u, 128, kb);
                v16bf bf = load_b_g_full(outw, 64, kb, nbase);
                acc = WMMA_BF16(a, bf, acc);
            }
            const int lane = tid & 31;
            const int N    = (lane & 15) + nbase;
            const int Mb   = (lane < 16) ? 0 : 8;
#pragma unroll
            for (int r = 0; r < 8; ++r)
                out[(brow + trow + Mb + r) * 64 + N] = acc[r];
        }
        __syncthreads();
    }

    // final states -> scratch for the FFT kernel
#pragma unroll
    for (int j = 0; j < 8; ++j)
        hout[(size_t)bg * 2048 + d_s * 16 + nb + j] = h[j];
}

// Separable 2D DFT (16-point along n, then 128-point along d), real part.
__global__ __launch_bounds__(256) void fft2_real_kernel(
    const float* __restrict__ hin, float* __restrict__ outp)
{
    __shared__ float sh[2048];
    __shared__ float sGr[2048];
    __shared__ float sGi[2048];
    const int bg  = blockIdx.x;
    const int tid = threadIdx.x;
    const float PI2 = 6.28318530717958647692f;

    const float* hp = hin + (size_t)bg * 2048;
    for (int i = tid; i < 2048; i += 256) sh[i] = hp[i];
    __syncthreads();

    // stage 1: G[d][n'] = sum_n h[d][n] * e^{-2*pi*i*n*n'/16}
    for (int i = tid * 8, e = tid * 8 + 8; i < e; ++i) {
        int d = i >> 4, np = i & 15;
        float gr = 0.0f, gi = 0.0f;
#pragma unroll
        for (int n = 0; n < 16; ++n) {
            int ph = (n * np) & 15;
            float th = PI2 * (float)ph * (1.0f / 16.0f);
            float hv = sh[d * 16 + n];
            gr += hv * __cosf(th);
            gi -= hv * __sinf(th);
        }
        sGr[i] = gr; sGi[i] = gi;
    }
    __syncthreads();

    // stage 2: Re(F[d'][n']) = sum_d Gr*cos + Gi*sin, theta = 2*pi*d*d'/128
    float* op = outp + (size_t)bg * 2048;
    for (int i = tid * 8, e = tid * 8 + 8; i < e; ++i) {
        int dp = i >> 4, np = i & 15;
        float re = 0.0f;
        for (int d = 0; d < 128; ++d) {
            int ph = (d * dp) & 127;
            float th = PI2 * (float)ph * (1.0f / 128.0f);
            re += sGr[d * 16 + np] * __cosf(th) + sGi[d * 16 + np] * __sinf(th);
        }
        op[i] = re;
    }
}

extern "C" void kernel_launch(void* const* d_in, const int* in_sizes, int n_in,
                              void* d_out, int out_size, void* d_ws, size_t ws_size,
                              hipStream_t stream) {
    const float* x      = (const float*)d_in[0];
    const float* latent = (const float*)d_in[1];
    const float* inw    = (const float*)d_in[2];
    const float* convw  = (const float*)d_in[3];
    const float* convb  = (const float*)d_in[4];
    const float* xpw    = (const float*)d_in[5];
    const float* dtw    = (const float*)d_in[6];
    const float* dtb    = (const float*)d_in[7];
    const float* outw   = (const float*)d_in[8];
    const float* alog   = (const float*)d_in[9];
    const float* Dp     = (const float*)d_in[10];

    float* out  = (float*)d_out;            // [8*4096*64] then [8*8*128*16]
    float* hbuf = (float*)d_ws;             // 64 * 2048 f32 = 512 KB scratch

    mamba_fused_scan<<<64, 256, 0, stream>>>(x, latent, inw, convw, convb, xpw,
                                             dtw, dtb, outw, alog, Dp, out, hbuf);
    fft2_real_kernel<<<64, 256, 0, stream>>>(hbuf, out + 8 * 4096 * 64);
}
